// SemanticCrossAttention_74096775790776
// MI455X (gfx1250) — matrix operand order repaired and checked by reading, not compiled, not run on hardware
//
#include <hip/hip_runtime.h>
#include <hip/hip_bf16.h>

// ---------- problem constants ----------
constexpr int BB    = 32;
constexpr int PP    = 8;
constexpr int KK    = 8;
constexpr int DD    = 768;        // feature dim
constexpr int GG    = 20000;      // genes
constexpr int TOPK  = 200;
constexpr int MQ    = BB * PP * KK;   // 2048 rows for Q projection
constexpr int MKF   = GG * KK;        // 160000 rows for Kf projection
constexpr int M2    = BB * PP;        // 256 rows for impact GEMM
constexpr int K2    = KK * DD;        // 6144 reduction dim for impact GEMM
constexpr float SCALE = 0.03608439182435161f;  // 768^-0.5

// ---------- vector types (trivial, union-safe) ----------
typedef __attribute__((ext_vector_type(16))) __bf16          v16bf;
typedef __attribute__((ext_vector_type(16))) unsigned short  v16u;
typedef __attribute__((ext_vector_type(8)))  float           v8f;
typedef __attribute__((ext_vector_type(4)))  unsigned int    v4u;

union BFrag {
    v16u         u;     // 16 bf16 bit patterns
    v16bf        b;     // view for the WMMA builtin
    v4u          q[2];  // two 16-byte chunks for vectorized bf16 loads
    unsigned int d[8];  // dword view for v_perm packing
};

// round-to-nearest-even fp32 -> bf16 (small epilogues only)
__device__ __forceinline__ unsigned short f2bf(float f) {
    unsigned int x = __builtin_bit_cast(unsigned int, f);
    unsigned int r = x + 0x7FFFu + ((x >> 16) & 1u);
    return (unsigned short)(r >> 16);
}

// pack two fp32 (raw bits) into one dword of two bf16 (truncation) via v_perm_b32
__device__ __forceinline__ unsigned int pack_bf16(unsigned int lo, unsigned int hi) {
    return __builtin_amdgcn_perm(hi, lo, 0x07060302u);
}

// ---------- kernel 0: softmax(w), conf_bias, emit w output ----------
__global__ void prep_kernel(const float* __restrict__ fw,
                            const float* __restrict__ conf,
                            float* __restrict__ ws_w,
                            float* __restrict__ ws_cb,
                            float* __restrict__ out_w) {
    __shared__ float w[KK];
    if (threadIdx.x == 0) {
        float m = fw[0];
        for (int k = 1; k < KK; ++k) m = fmaxf(m, fw[k]);
        float e[KK]; float s = 0.f;
        for (int k = 0; k < KK; ++k) { e[k] = expf(fw[k] - m); s += e[k]; }
        float inv = 1.f / s;
        for (int k = 0; k < KK; ++k) {
            w[k] = e[k] * inv;
            ws_w[k]  = w[k];
            out_w[k] = w[k];
        }
    }
    __syncthreads();
    int t = threadIdx.x;               // one (b,p) per thread, 256 threads
    float cb = 0.f;
    for (int k = 0; k < KK; ++k) {
        float c = fmaxf(conf[t * KK + k], 1e-6f);
        cb += logf(c) * w[k];
    }
    ws_cb[t] = cb;
}

// ---------- kernel 1/2: P = X @ W^T + b  (fp32 in, bf16 out, optional row scale) ----------
// One wave computes a 64x16 output tile (4 M-tiles). Per K-step: build B fragment once,
// build all 4 A fragments, then fire 4 independent WMMAs back-to-back (no intra-burst
// register hazards -> hazard nops only at iteration boundaries).
// grid = (DD/16, Mrows/256), block = 128 (4 waves).
__global__ void proj_kernel(const float* __restrict__ X,
                            const float* __restrict__ W,
                            const float* __restrict__ bias,
                            const float* __restrict__ wsw,
                            unsigned short* __restrict__ out,
                            int Mrows, int useRowScale) {
    const int lane   = threadIdx.x & 31;
    const int wave   = threadIdx.x >> 5;
    const int nBase  = blockIdx.x * 16;
    const int mBlock = blockIdx.y * 256 + wave * 64;   // this wave's 4 M-tiles
    if (mBlock >= Mrows) return;
    const int half = lane >> 4;
    const int l16  = lane & 15;

    const unsigned int* bBase =
        (const unsigned int*)(W + (size_t)(nBase + l16) * DD);
    const unsigned int* aBase[4];
#pragma unroll
    for (int t = 0; t < 4; ++t)
        aBase[t] = (const unsigned int*)(X + (size_t)(mBlock + t * 16 + l16) * DD);

    v8f acc[4] = {};
    for (int kb = 0; kb < DD; kb += 32) {
        // B fragment: element i <-> K = 16*half + i, N = lane%16; B(K=d,N=n) = W[n*DD+d]
        unsigned int ub[16];
        *(v4u*)(ub + 0)  = *(const v4u*)(bBase + kb + half * 16);
        *(v4u*)(ub + 4)  = *(const v4u*)(bBase + kb + half * 16 + 4);
        *(v4u*)(ub + 8)  = *(const v4u*)(bBase + kb + half * 16 + 8);
        *(v4u*)(ub + 12) = *(const v4u*)(bBase + kb + half * 16 + 12);
        BFrag fb;
#pragma unroll
        for (int j = 0; j < 8; ++j) fb.d[j] = pack_bf16(ub[2 * j], ub[2 * j + 1]);

        // All 4 A fragments first ...
        BFrag fa[4];
#pragma unroll
        for (int t = 0; t < 4; ++t) {
            // A fragment: lanes 0-15 -> K {0..15}, lanes 16-31 -> K {16..31}
            unsigned int ua[16];
            *(v4u*)(ua + 0)  = *(const v4u*)(aBase[t] + kb + half * 16);
            *(v4u*)(ua + 4)  = *(const v4u*)(aBase[t] + kb + half * 16 + 4);
            *(v4u*)(ua + 8)  = *(const v4u*)(aBase[t] + kb + half * 16 + 8);
            *(v4u*)(ua + 12) = *(const v4u*)(aBase[t] + kb + half * 16 + 12);
            __builtin_prefetch(aBase[t] + kb + 128, 0, 1);   // stream X rows
#pragma unroll
            for (int j = 0; j < 8; ++j) fa[t].d[j] = pack_bf16(ua[2 * j], ua[2 * j + 1]);
        }
        // ... then a burst of 4 independent WMMAs (shared B, distinct A/accum)
#pragma unroll
        for (int t = 0; t < 4; ++t)
            acc[t] = __builtin_amdgcn_wmma_f32_16x16x32_bf16(
                false, fa[t].b, false, fb.b, (short)0, acc[t], false, false);
    }

    // Epilogue: row m = mBlock + t*16 + v + 8*half, mBlock % 16 == 0 -> m % KK == v,
    // so the facet weight per accumulator element is uniform; branch-free.
    float rs8[8];
#pragma unroll
    for (int v = 0; v < 8; ++v) rs8[v] = 1.0f;
    if (useRowScale) {
#pragma unroll
        for (int v = 0; v < 8; ++v) rs8[v] = SCALE * wsw[v];
    }
    const float bn = bias[nBase + l16];
#pragma unroll
    for (int t = 0; t < 4; ++t) {
#pragma unroll
        for (int v = 0; v < 8; ++v) {
            int m = mBlock + t * 16 + v + half * 8;
            float val = (acc[t][v] + bn) * rs8[v];
            out[(size_t)m * DD + nBase + l16] = f2bf(val);
        }
    }
}

// ---------- kernel 3: impact = Qw(256xK2) @ Kf(20000xK2)^T + conf_bias ----------
// One wave computes 64x16 (4 M-tiles); block of 4 waves covers all 256 A-rows, so
// each Kf (B) row streams ~once from HBM, with the 4x in-block reuse hitting L1/L2.
// grid = (GG/16), block = 128.
__global__ void impact_kernel(const unsigned short* __restrict__ Qw,
                              const unsigned short* __restrict__ Kf,
                              const float* __restrict__ cb,
                              float* __restrict__ impact) {
    const int lane  = threadIdx.x & 31;
    const int wave  = threadIdx.x >> 5;
    const int nBase = blockIdx.x * 16;
    const int half  = lane >> 4;
    const int l16   = lane & 15;

    const unsigned short* bBase = Kf + (size_t)(nBase + l16) * K2;
    const unsigned short* aBase[4];
#pragma unroll
    for (int t = 0; t < 4; ++t)
        aBase[t] = Qw + (size_t)(wave * 64 + t * 16 + l16) * K2;

    v8f acc[4] = {};
    for (int kb = 0; kb < K2; kb += 32) {
        BFrag fb;
        fb.q[0] = *(const v4u*)(bBase + kb + half * 16);
        fb.q[1] = *(const v4u*)(bBase + kb + half * 16 + 8);
        __builtin_prefetch(bBase + kb + 512, 0, 1);    // stream Kf
        BFrag fa[4];
#pragma unroll
        for (int t = 0; t < 4; ++t) {
            fa[t].q[0] = *(const v4u*)(aBase[t] + kb + half * 16);
            fa[t].q[1] = *(const v4u*)(aBase[t] + kb + half * 16 + 8);
        }
#pragma unroll
        for (int t = 0; t < 4; ++t)
            acc[t] = __builtin_amdgcn_wmma_f32_16x16x32_bf16(
                false, fa[t].b, false, fb.b, (short)0, acc[t], false, false);
    }
#pragma unroll
    for (int t = 0; t < 4; ++t) {
#pragma unroll
        for (int v = 0; v < 8; ++v) {
            int m = wave * 64 + t * 16 + v + half * 8;
            impact[(size_t)m * GG + nBase + l16] = acc[t][v] + cb[m];
        }
    }
}

// ---------- kernel 4: per-row top-200 mask via count bisection ----------
__global__ void topk_kernel(const float* __restrict__ impact,
                            float* __restrict__ out) {
    const int row = blockIdx.x;                        // 0..255
    const float* src = impact + (size_t)row * GG;
    float* dst = out + (size_t)row * GG;
    __shared__ float red[256];
    __shared__ int cnt;

    float lmax = -3.4e38f, lmin = 3.4e38f;
    for (int i = threadIdx.x; i < GG; i += 256) {
        float v = src[i];
        lmax = fmaxf(lmax, v); lmin = fminf(lmin, v);
    }
    red[threadIdx.x] = lmax; __syncthreads();
    for (int s = 128; s > 0; s >>= 1) {
        if (threadIdx.x < s) red[threadIdx.x] = fmaxf(red[threadIdx.x], red[threadIdx.x + s]);
        __syncthreads();
    }
    float hi = red[0]; __syncthreads();
    red[threadIdx.x] = lmin; __syncthreads();
    for (int s = 128; s > 0; s >>= 1) {
        if (threadIdx.x < s) red[threadIdx.x] = fminf(red[threadIdx.x], red[threadIdx.x + s]);
        __syncthreads();
    }
    float lo = red[0]; __syncthreads();

    for (int it = 0; it < 32; ++it) {
        float mid = 0.5f * (lo + hi);
        if (threadIdx.x == 0) cnt = 0;
        __syncthreads();
        int c = 0;
        for (int i = threadIdx.x; i < GG; i += 256) c += (src[i] > mid) ? 1 : 0;
        atomicAdd(&cnt, c);
        __syncthreads();
        int total = cnt;
        if (total > TOPK) lo = mid; else hi = mid;     // uniform decision across block
        __syncthreads();
    }
    const float thr = hi;                              // count(> thr) <= TOPK (~= TOPK)
    for (int i = threadIdx.x; i < GG; i += 256) {
        float v = src[i];
        dst[i] = (v > thr) ? v : 0.0f;
    }
}

// ---------- host-side launch ----------
extern "C" void kernel_launch(void* const* d_in, const int* in_sizes, int n_in,
                              void* d_out, int out_size, void* d_ws, size_t ws_size,
                              hipStream_t stream) {
    const float* ctxQ   = (const float*)d_in[0];   // (B,P,K,D)
    const float* genome = (const float*)d_in[1];   // (G,K,D)
    const float* conf   = (const float*)d_in[2];   // (B,P,K)
    const float* Wq     = (const float*)d_in[3];   // (D,D)
    const float* bq     = (const float*)d_in[4];   // (D,)
    const float* Wk     = (const float*)d_in[5];   // (D,D)
    const float* bk     = (const float*)d_in[6];   // (D,)
    const float* fw     = (const float*)d_in[7];   // (K,)
    float* out = (float*)d_out;                    // impact (B,P,G) then w (K,)

    // workspace layout
    char* ws = (char*)d_ws;
    constexpr size_t OFF_W   = 0;                                   // 8 f32
    constexpr size_t OFF_CB  = 256;                                 // 256 f32
    constexpr size_t OFF_QW  = 4096;                                // 2048*768 bf16
    constexpr size_t QW_B    = (size_t)MQ * DD * 2;                 // 3,145,728
    constexpr size_t OFF_KF  = OFF_QW + QW_B;                       // 160000*768 bf16
    constexpr size_t KF_B    = (size_t)MKF * DD * 2;                // 245,760,000
    constexpr size_t OFF_IMP = OFF_KF + KF_B;                       // 256*20000 f32

    float*          ws_w   = (float*)(ws + OFF_W);
    float*          ws_cb  = (float*)(ws + OFF_CB);
    unsigned short* ws_Qw  = (unsigned short*)(ws + OFF_QW);
    unsigned short* ws_Kf  = (unsigned short*)(ws + OFF_KF);
    float*          ws_imp = (float*)(ws + OFF_IMP);

    // 0) softmax weights + confidence bias (+ emit w output)
    prep_kernel<<<1, 256, 0, stream>>>(fw, conf, ws_w, ws_cb,
                                       out + (size_t)BB * PP * GG);
    // 1) Qw = (ctxQ @ Wq^T + bq) * scale * w[k]   -> bf16
    proj_kernel<<<dim3(DD / 16, MQ / 256), 128, 0, stream>>>(
        ctxQ, Wq, bq, ws_w, ws_Qw, MQ, 1);
    // 2) Kf = genome @ Wk^T + bk                  -> bf16
    proj_kernel<<<dim3(DD / 16, MKF / 256), 128, 0, stream>>>(
        genome, Wk, bk, ws_w, ws_Kf, MKF, 0);
    // 3) impact = Qw @ Kf^T + conf_bias           -> f32 (ws)
    impact_kernel<<<dim3(GG / 16), 128, 0, stream>>>(
        ws_Qw, ws_Kf, ws_cb, ws_imp);
    // 4) top-200 mask per (b,p) row               -> d_out
    topk_kernel<<<M2, 256, 0, stream>>>(ws_imp, out);
}
